// BasicBlock_Q_35356170781048
// MI455X (gfx1250) — matrix operand order repaired and checked by reading, not compiled, not run on hardware
//
#include <hip/hip_runtime.h>
#include <hip/hip_bf16.h>
#include <math.h>

typedef __attribute__((ext_vector_type(16))) _Float16 v16h;
typedef __attribute__((ext_vector_type(8)))  float    v8f;

#define CCH    64           // channels
#define HW     64           // 8x8 spatial
#define NIMG   2048
#define KTOT   576          // 9 taps * 64 channels, K-permuted: k' = r*64 + ci
#define KSTEPS 18           // 576 / 32
#define LDS_STRIDE 80       // halves per padded spatial position (160 B, 32B-aligned frags)
#define NT_PER_CH 131072.0f // NIMG * HW

// ---------------------------------------------------------------------------
// 0) zero BN statistic accumulators (256 floats) every call
// ---------------------------------------------------------------------------
__global__ void zero_stats_kernel(float* stats) {
  stats[threadIdx.x] = 0.0f;
}

// ---------------------------------------------------------------------------
// 1) max|w| -> tanh(max|w|) == max|tanh(w)| (tanh odd & monotone)
// ---------------------------------------------------------------------------
__global__ void wmax_kernel(const float* __restrict__ w, float* __restrict__ out) {
  __shared__ float red[256];
  float m = 0.0f;
  for (int i = threadIdx.x; i < CCH * CCH * 9; i += 256)
    m = fmaxf(m, fabsf(w[i]));
  red[threadIdx.x] = m;
  __syncthreads();
  for (int s = 128; s > 0; s >>= 1) {
    if (threadIdx.x < s) red[threadIdx.x] = fmaxf(red[threadIdx.x], red[threadIdx.x + s]);
    __syncthreads();
  }
  if (threadIdx.x == 0) out[0] = tanhf(red[0]);
}

// ---------------------------------------------------------------------------
// 2) DoReFa weight quant + mask, packed into WMMA A-fragment order (f16),
//    with the K-permutation k' = r*64 + ci  (tap-major, channel-minor).
//    Fragment index: ((s*4 + t)*32 + lane)*16 + e, contiguous K per lane half (same as B frag):
//    lanes 0-15: K=0..15 ; lanes 16-31: K=16..31.
// ---------------------------------------------------------------------------
__global__ void pack_weights_kernel(const float* __restrict__ w,
                                    const float* __restrict__ pat,
                                    const float* __restrict__ wmax,
                                    _Float16* __restrict__ frag) {
  int tid = blockIdx.x * 256 + threadIdx.x;   // (s, t, lane)
  if (tid >= KSTEPS * 4 * 32) return;
  int lane = tid & 31;
  int t    = (tid >> 5) & 3;
  int s    = tid >> 7;
  int m    = t * 16 + (lane & 15);
  int half = lane >> 4;
  float inv = 1.0f / (2.0f * wmax[0]);
  _Float16* dst = frag + (size_t)tid * 16;
#pragma unroll
  for (int e = 0; e < 16; ++e) {
    int klocal = half * 16 + e;
    int k  = s * 32 + klocal;                 // permuted K index
    int r  = k >> 6;                          // tap 0..8  (ky*3+kx)
    int ci = k & 63;                          // channel
    int idx = (m * CCH + ci) * 9 + r;         // OIHW flat
    float tv  = tanhf(w[idx]);
    float w01 = tv * inv + 0.5f;
    float q   = rintf(w01 * 15.0f) * (1.0f / 15.0f);
    dst[e] = (_Float16)((2.0f * q - 1.0f) * pat[idx]);
  }
}

// ---------------------------------------------------------------------------
// 3) implicit-GEMM conv3x3, one workgroup (8 wave32) per image.
//    LDS input tile: channel-minor s_in[pos*80 + ci] (10x10 padded), so each
//    lane's B fragment (one tap, 16 consecutive channels) is ONE aligned
//    32-byte LDS read. 18 k-steps x 2 v_wmma_f32_16x16x32_f16 per wave.
//    Output stored channel-minor [n][pos][c] -> one v8f store per tile.
//    BN stats: 16-lane shfl butterfly -> LDS atomics -> 1 global atomic/ch.
// ---------------------------------------------------------------------------
template <typename TIN, bool POSMAJOR>
__global__ __launch_bounds__(256) void conv_gemm_kernel(
    const TIN* __restrict__ in, const _Float16* __restrict__ wfrag,
    float* __restrict__ out, float* __restrict__ gsum, float* __restrict__ gssq) {
  __shared__ _Float16 s_in[100 * LDS_STRIDE] __attribute__((aligned(32)));
  __shared__ float s_sum[CCH], s_ssq[CCH];

  const int tid = threadIdx.x;
  const int n_img = blockIdx.x;

  for (int i = tid; i < 100 * LDS_STRIDE; i += 256) s_in[i] = (_Float16)0.0f;
  if (tid < CCH) { s_sum[tid] = 0.0f; s_ssq[tid] = 0.0f; }
  __syncthreads();

  const TIN* src = in + (size_t)n_img * (CCH * HW);
  for (int i = tid; i < CCH * HW; i += 256) {
    int ci, p;
    if (POSMAJOR) { p = i >> 6; ci = i & 63; }      // [pos][ch] source
    else          { ci = i >> 6; p = i & 63; }      // [ch][pos] source (NCHW)
    int y = p >> 3, x = p & 7;
    s_in[((y + 1) * 10 + (x + 1)) * LDS_STRIDE + ci] = (_Float16)(float)src[i];
  }
  __syncthreads();

  const int wave = tid >> 5, lane = tid & 31;
  const int n_tile = wave & 3;
  const int t0 = (wave >> 2) * 2, t1 = t0 + 1;
  const int ncol = n_tile * 16 + (lane & 15);      // output spatial position
  const int oy = ncol >> 3, ox = ncol & 7;
  const int half = lane >> 4;
  const int kbase = half * 16;                     // B-frag K base within step

  v8f acc0 = {}; v8f acc1 = {};
  const v16h* wf = (const v16h*)wfrag;

  for (int s = 0; s < KSTEPS; ++s) {
    int kb = s * 32 + kbase;                       // multiple of 16
    int r  = kb >> 6;                              // tap (constant over frag)
    int cb = kb & 63;                              // first channel of frag
    int ky = r / 3, kx = r - 3 * ky;
    const v16h b = *(const v16h*)(s_in + ((oy + ky) * 10 + (ox + kx)) * LDS_STRIDE + cb);
    v16h a0 = wf[(s * 4 + t0) * 32 + lane];
    v16h a1 = wf[(s * 4 + t1) * 32 + lane];
    if (s + 1 < KSTEPS) {
      __builtin_prefetch((const void*)(wf + ((s + 1) * 4 + t0) * 32 + lane), 0, 3);
      __builtin_prefetch((const void*)(wf + ((s + 1) * 4 + t1) * 32 + lane), 0, 3);
    }
    acc0 = __builtin_amdgcn_wmma_f32_16x16x32_f16(false, a0, false, b,
                                                  (short)0, acc0, false, false);
    acc1 = __builtin_amdgcn_wmma_f32_16x16x32_f16(false, a1, false, b,
                                                  (short)0, acc1, false, false);
  }

  // C/D layout: VGPR v -> row m = t*16 + half*8 + v, col = lane&15 -> ncol.
  // Channel-minor store: 8 consecutive channels = one aligned v8f store.
  float* obase = out + (size_t)n_img * (CCH * HW) + ncol * CCH;
  *(v8f*)(obase + t0 * 16 + half * 8) = acc0;
  *(v8f*)(obase + t1 * 16 + half * 8) = acc1;

  // BN stats: reduce over the 16 columns sharing this channel set.
#pragma unroll
  for (int v = 0; v < 8; ++v) {
    float v0 = acc0[v], q0 = v0 * v0;
    float v1 = acc1[v], q1 = v1 * v1;
#pragma unroll
    for (int mask = 1; mask < 16; mask <<= 1) {
      v0 += __shfl_xor(v0, mask, 16); q0 += __shfl_xor(q0, mask, 16);
      v1 += __shfl_xor(v1, mask, 16); q1 += __shfl_xor(q1, mask, 16);
    }
    if ((lane & 15) == 0) {
      int m0 = t0 * 16 + half * 8 + v;
      int m1 = t1 * 16 + half * 8 + v;
      atomicAdd(&s_sum[m0], v0); atomicAdd(&s_ssq[m0], q0);
      atomicAdd(&s_sum[m1], v1); atomicAdd(&s_ssq[m1], q1);
    }
  }
  __syncthreads();
  if (tid < CCH) {
    atomicAdd(&gsum[tid], s_sum[tid]);
    atomicAdd(&gssq[tid], s_ssq[tid]);
  }
}

// ---------------------------------------------------------------------------
// 4) BN finalize: per-channel scale/shift
// ---------------------------------------------------------------------------
__global__ void bn_finalize_kernel(const float* __restrict__ gsum,
                                   const float* __restrict__ gssq,
                                   const float* __restrict__ gamma,
                                   const float* __restrict__ beta,
                                   float* __restrict__ scale,
                                   float* __restrict__ shift) {
  int c = threadIdx.x;
  if (c < CCH) {
    float mean = gsum[c] * (1.0f / NT_PER_CH);
    float var  = gssq[c] * (1.0f / NT_PER_CH) - mean * mean;
    float sc   = gamma[c] * rsqrtf(var + 1e-5f);
    scale[c] = sc;
    shift[c] = beta[c] - mean * sc;
  }
}

// ---------------------------------------------------------------------------
// 5) BN apply + 4-bit act quant -> f16 activation for conv2.
//    conv-out and act1 are both [n][pos][c] channel-minor: pure elementwise.
// ---------------------------------------------------------------------------
__global__ void bn_quant_kernel(const float* __restrict__ in,
                                const float* __restrict__ scale,
                                const float* __restrict__ shift,
                                _Float16* __restrict__ outq) {
  size_t i = (size_t)blockIdx.x * 256 + threadIdx.x;
  int c = (int)(i & 63);
  float v = in[i] * scale[c] + shift[c];
  v = fminf(fmaxf(v, 0.0f), 1.0f);
  outq[i] = (_Float16)(rintf(v * 15.0f) * (1.0f / 15.0f));
}

// ---------------------------------------------------------------------------
// 6) BN2 + residual + act quant -> f32 output in reference NCHW layout.
//    Per-image LDS transpose (pad 65) so all global traffic is coalesced.
// ---------------------------------------------------------------------------
__global__ __launch_bounds__(256) void bn_residual_quant_kernel(
    const float* __restrict__ conv,      // [n][pos][c]
    const float* __restrict__ x,         // [n][c][pos] (NCHW)
    const float* __restrict__ scale,
    const float* __restrict__ shift,
    float* __restrict__ out) {           // [n][c][pos] (NCHW)
  __shared__ float tile[CCH * 65];
  const size_t base = (size_t)blockIdx.x * (CCH * HW);
  for (int i = threadIdx.x; i < CCH * HW; i += 256) {
    int p = i >> 6, c = i & 63;
    tile[c * 65 + p] = conv[base + i] * scale[c] + shift[c];
  }
  __syncthreads();
  for (int j = threadIdx.x; j < CCH * HW; j += 256) {
    int c = j >> 6, p = j & 63;
    float v = tile[c * 65 + p] + x[base + j];
    v = fminf(fmaxf(v, 0.0f), 1.0f);
    out[base + j] = rintf(v * 15.0f) * (1.0f / 15.0f);
  }
}

// ---------------------------------------------------------------------------
extern "C" void kernel_launch(void* const* d_in, const int* in_sizes, int n_in,
                              void* d_out, int out_size, void* d_ws, size_t ws_size,
                              hipStream_t stream) {
  const float* x      = (const float*)d_in[0];
  const float* w1     = (const float*)d_in[1];
  const float* w2     = (const float*)d_in[2];
  const float* pat1   = (const float*)d_in[3];
  const float* pat2   = (const float*)d_in[4];
  const float* gamma1 = (const float*)d_in[5];
  const float* beta1  = (const float*)d_in[6];
  const float* gamma2 = (const float*)d_in[7];
  const float* beta2  = (const float*)d_in[8];
  float* out = (float*)d_out;

  // workspace layout
  char* ws = (char*)d_ws;
  float* wmax1 = (float*)(ws + 0);
  float* wmax2 = wmax1 + 1;
  float* stats = (float*)(ws + 256);          // 256 floats
  float* sum1  = stats + 0,   *ssq1 = stats + 64;
  float* sum2  = stats + 128, *ssq2 = stats + 192;
  float* bnp   = (float*)(ws + 1536);         // 256 floats
  float* scale1 = bnp + 0,   *shift1 = bnp + 64;
  float* scale2 = bnp + 128, *shift2 = bnp + 192;
  _Float16* wfrag1 = (_Float16*)(ws + 4096);                 // 73728 B
  _Float16* wfrag2 = (_Float16*)(ws + 4096 + 73728);         // 73728 B
  _Float16* act1   = (_Float16*)(ws + 4096 + 2 * 73728);     // 16 MiB, [n][pos][c]
  float*    out1   = (float*)(ws + 4096 + 2 * 73728 + 16777216); // 32 MiB, reused
  float*    out2   = out1;

  const int NELEM_BLOCKS = (NIMG * CCH * HW) / 256;  // 32768

  zero_stats_kernel<<<1, 256, 0, stream>>>(stats);
  wmax_kernel<<<1, 256, 0, stream>>>(w1, wmax1);
  wmax_kernel<<<1, 256, 0, stream>>>(w2, wmax2);
  pack_weights_kernel<<<9, 256, 0, stream>>>(w1, pat1, wmax1, wfrag1);
  pack_weights_kernel<<<9, 256, 0, stream>>>(w2, pat2, wmax2, wfrag2);

  conv_gemm_kernel<float, false><<<NIMG, 256, 0, stream>>>(x, wfrag1, out1, sum1, ssq1);
  bn_finalize_kernel<<<1, 64, 0, stream>>>(sum1, ssq1, gamma1, beta1, scale1, shift1);
  bn_quant_kernel<<<NELEM_BLOCKS, 256, 0, stream>>>(out1, scale1, shift1, act1);

  conv_gemm_kernel<_Float16, true><<<NIMG, 256, 0, stream>>>(act1, wfrag2, out2, sum2, ssq2);
  bn_finalize_kernel<<<1, 64, 0, stream>>>(sum2, ssq2, gamma2, beta2, scale2, shift2);
  bn_residual_quant_kernel<<<NIMG, 256, 0, stream>>>(out2, x, scale2, shift2, out);
}
